// FastMambaBlock_9904194584654
// MI455X (gfx1250) — compile-verified, matrix operand order checked by reading, not run on hardware
//
#include <hip/hip_runtime.h>
#include <hip/hip_bf16.h>
#include <cstdint>

// ---------------------------------------------------------------------------
// Mamba block for MI455X (gfx1250, wave32, WMMA).
// GEMMs: bf16 WMMA (v_wmma_f32_16x16x32_bf16), f32 accumulate,
//        double-buffered LDS staged via global_load_async_to_lds_b128
//        (ASYNCcnt-tracked DMA, no VGPR round trip).
// Scan:  16 lanes per (b,d) chain (one lane per state), shfl_xor reduce.
// Workspace requirement: ~354 MB.
// ---------------------------------------------------------------------------

#define D_MODEL 1024
#define D_STATE 16
#define D_CONV 4
#define D_INNER 2048
#define DT_RANK 64
#define BATCH 4
#define SEQLEN 2048
#define NROWS (BATCH * SEQLEN)          // 8192 flattened (b,l) rows
#define XDBL_N (DT_RANK + 2 * D_STATE)  // 96

typedef __attribute__((ext_vector_type(16))) __bf16 v16bf;
typedef __attribute__((ext_vector_type(8)))  __bf16 v8bf;
typedef __attribute__((ext_vector_type(8)))  float  v8f;

// ---------------------------------------------------------------------------
// gfx1250 async global->LDS DMA (ASYNCcnt).  lds_off = low 32 bits of the
// flat shared-memory address.
// ---------------------------------------------------------------------------
__device__ __forceinline__ void async_load_b128(uint32_t lds_off, const void* gptr) {
  asm volatile("global_load_async_to_lds_b128 %0, %1, off"
               :: "v"(lds_off), "v"(gptr)
               : "memory");
}
__device__ __forceinline__ void wait_async_le6() {
  asm volatile("s_wait_asynccnt 0x6" ::: "memory");
}
__device__ __forceinline__ void wait_async_0() {
  asm volatile("s_wait_asynccnt 0x0" ::: "memory");
}

// ---------------------------------------------------------------------------
// f32 -> bf16 conversion (weights)
// ---------------------------------------------------------------------------
__global__ __launch_bounds__(256) void f32_to_bf16_kernel(
    const float* __restrict__ src, __bf16* __restrict__ dst, int n) {
  int i = blockIdx.x * 256 + threadIdx.x;
  if (i < n) dst[i] = (__bf16)src[i];
}

// ---------------------------------------------------------------------------
// Fused LayerNorm over D_MODEL, emits bf16 for GEMM1.  One block per row.
// ---------------------------------------------------------------------------
__global__ __launch_bounds__(256) void layernorm_bf16_kernel(
    const float* __restrict__ x, const float* __restrict__ gamma,
    const float* __restrict__ beta, __bf16* __restrict__ out) {
  const int row = blockIdx.x;
  const float* xr = x + (size_t)row * D_MODEL;
  float v[4];
  float s = 0.f, ss = 0.f;
#pragma unroll
  for (int i = 0; i < 4; ++i) {
    v[i] = xr[threadIdx.x + i * 256];
    s += v[i];
    ss += v[i] * v[i];
  }
#pragma unroll
  for (int o = 16; o > 0; o >>= 1) {
    s += __shfl_down(s, o);
    ss += __shfl_down(ss, o);
  }
  __shared__ float sh_s[8], sh_ss[8];
  __shared__ float sh_mu, sh_ri;
  const int w = threadIdx.x >> 5;
  if ((threadIdx.x & 31) == 0) { sh_s[w] = s; sh_ss[w] = ss; }
  __syncthreads();
  if (threadIdx.x == 0) {
    float ts = 0.f, tss = 0.f;
#pragma unroll
    for (int i = 0; i < 8; ++i) { ts += sh_s[i]; tss += sh_ss[i]; }
    float mu = ts * (1.f / D_MODEL);
    float var = tss * (1.f / D_MODEL) - mu * mu;
    sh_mu = mu;
    sh_ri = rsqrtf(var + 1e-5f);
  }
  __syncthreads();
  const float mu = sh_mu, ri = sh_ri;
#pragma unroll
  for (int i = 0; i < 4; ++i) {
    int c = threadIdx.x + i * 256;
    out[(size_t)row * D_MODEL + c] = (__bf16)(((v[i] - mu) * ri) * gamma[c] + beta[c]);
  }
}

// ---------------------------------------------------------------------------
// bf16 WMMA GEMM:  C[M,N] = A[M,K] * B[N,K]^T   (both operands K-major)
// Block: 256 threads = 8 waves, tile 128x64, BK=64 (two WMMA K-steps/stage).
// Wave grid 4x2, each wave owns 32x32 = 2x2 WMMA tiles.
// Double-buffered LDS, filled with async global->LDS b128 DMA (6 per wave
// per stage: 4 for A, 2 for B; uniform count so s_wait_asynccnt 6 overlaps
// stage i+1 DMA with stage i WMMA).
// ---------------------------------------------------------------------------
#define GBM 128
#define GBN 64
#define GBK 64
#define GLDT (GBK + 8)          // padded LDS row, bf16 elems (144 bytes)
#define GROWB (GLDT * 2)        // row stride bytes

__device__ __forceinline__ v16bf load_frag_row(const __bf16* rowp, int lk) {
  // lane holds 8 contiguous K at lk and 8 at lk+16 (CDNA5 16-bit A/B layout)
  v8bf lo = *reinterpret_cast<const v8bf*>(rowp + lk);
  v8bf hi = *reinterpret_cast<const v8bf*>(rowp + lk + 16);
  v16bf r;
#pragma unroll
  for (int i = 0; i < 8; ++i) { r[i] = lo[i]; r[i + 8] = hi[i]; }
  return r;
}

template <bool ADD_RES>
__global__ __launch_bounds__(256) void gemm_bf16_wmma(
    const __bf16* __restrict__ A, const __bf16* __restrict__ Bw,
    float* __restrict__ C, const float* __restrict__ res,
    int M, int N, int K) {
  __shared__ __align__(16) __bf16 sA[2][GBM * GLDT];
  __shared__ __align__(16) __bf16 sB[2][GBN * GLDT];

  const int tid = threadIdx.x;
  const int lane = tid & 31;
  const int wave = tid >> 5;
  const int wm = wave >> 1;        // 0..3
  const int wn = wave & 1;         // 0..1
  const int lm = lane & 15;
  const int lk = (lane >> 4) * 8;  // 0 or 8

  const int blockRow = blockIdx.y * GBM;
  const int blockCol = blockIdx.x * GBN;

  // Per-thread DMA chunk geometry (16-byte chunks).
  // A tile: 128 rows x 128B = 1024 chunks -> 4/thread.
  // B tile:  64 rows x 128B =  512 chunks -> 2/thread (rows clamped to N-1;
  // garbage feeds only output cols >= N which are never stored).
  int aRow[4], aCb[4];
#pragma unroll
  for (int i = 0; i < 4; ++i) {
    int chunk = tid + i * 256;
    aRow[i] = chunk >> 3;
    aCb[i] = (chunk & 7) * 16;
  }
  int bRow[2], bCb[2];
#pragma unroll
  for (int i = 0; i < 2; ++i) {
    int chunk = tid + i * 256;
    int r = chunk >> 3;
    int rr = blockCol + r;
    bRow[i] = (rr < N) ? rr : (N - 1);
    bCb[i] = (chunk & 7) * 16;
    bRow[i] = bRow[i] - blockCol;            // local row for LDS placement
    // keep the clamped global row separately:
    // (recomputed below from bRowG)
  }
  int bRowG[2];
#pragma unroll
  for (int i = 0; i < 2; ++i) {
    int chunk = tid + i * 256;
    int r = chunk >> 3;
    int rr = blockCol + r;
    bRowG[i] = (rr < N) ? rr : (N - 1);
    bRow[i] = r;                              // LDS row = tile row
  }

  auto stage = [&](int buf, int k0) {
    const char* aBase = (const char*)A + (size_t)k0 * 2;
    const char* bBase = (const char*)Bw + (size_t)k0 * 2;
    char* la = (char*)&sA[buf][0];
    char* lb = (char*)&sB[buf][0];
#pragma unroll
    for (int i = 0; i < 4; ++i) {
      const char* g = aBase + (size_t)(blockRow + aRow[i]) * K * 2 + aCb[i];
      uint32_t l = (uint32_t)(uintptr_t)(la + aRow[i] * GROWB + aCb[i]);
      async_load_b128(l, g);
    }
#pragma unroll
    for (int i = 0; i < 2; ++i) {
      const char* g = bBase + (size_t)bRowG[i] * K * 2 + bCb[i];
      uint32_t l = (uint32_t)(uintptr_t)(lb + bRow[i] * GROWB + bCb[i]);
      async_load_b128(l, g);
    }
  };

  v8f acc[2][2] = {};

  int cur = 0;
  stage(0, 0);
  for (int k0 = 0; k0 < K; k0 += GBK) {
    const bool more = (k0 + GBK) < K;
    if (more) {
      stage(cur ^ 1, k0 + GBK);
      wait_async_le6();   // previous stage's 6 DMAs complete (in-order)
    } else {
      wait_async_0();
    }
    __syncthreads();

#pragma unroll
    for (int ks = 0; ks < 2; ++ks) {
      v16bf af[2], bfv[2];
#pragma unroll
      for (int i = 0; i < 2; ++i)
        af[i] = load_frag_row(&sA[cur][(wm * 32 + i * 16 + lm) * GLDT + ks * 32], lk);
#pragma unroll
      for (int j = 0; j < 2; ++j)
        bfv[j] = load_frag_row(&sB[cur][(wn * 32 + j * 16 + lm) * GLDT + ks * 32], lk);
#pragma unroll
      for (int i = 0; i < 2; ++i)
#pragma unroll
        for (int j = 0; j < 2; ++j)
          acc[i][j] = __builtin_amdgcn_wmma_f32_16x16x32_bf16(
              false, af[i], false, bfv[j], (short)0, acc[i][j], false, false);
    }

    __syncthreads();  // all waves done reading buf[cur] before it is re-filled
    cur ^= 1;
  }

  // C/D layout: VGPR r on lane l -> (m = r + (l>>4)*8, n = l&15)
#pragma unroll
  for (int i = 0; i < 2; ++i) {
#pragma unroll
    for (int j = 0; j < 2; ++j) {
      int row0 = blockRow + wm * 32 + i * 16 + (lane >> 4) * 8;
      int col = blockCol + wn * 32 + j * 16 + lm;
      if (col < N) {
#pragma unroll
        for (int r = 0; r < 8; ++r) {
          size_t off = (size_t)(row0 + r) * N + col;
          float v = acc[i][j][r];
          if (ADD_RES) v += res[off];
          C[off] = v;
        }
      }
    }
  }
}

// ---------------------------------------------------------------------------
// Depthwise causal conv (taps=4) + bias + SiLU.  Reads xs half of xz.
// ---------------------------------------------------------------------------
__global__ __launch_bounds__(256) void conv_silu_kernel(
    const float* __restrict__ xz, const float* __restrict__ cw,
    const float* __restrict__ cb, float* __restrict__ xc,
    __bf16* __restrict__ xcb) {
  size_t idx = (size_t)blockIdx.x * 256 + threadIdx.x;  // over NROWS*D_INNER
  int d = (int)(idx & (D_INNER - 1));
  size_t bl = idx >> 11;
  int l = (int)(bl & (SEQLEN - 1));
  int b = (int)(bl >> 11);
  float acc = cb[d];
#pragma unroll
  for (int j = 0; j < D_CONV; ++j) {
    int li = l - (D_CONV - 1) + j;
    if (li >= 0)
      acc += xz[((size_t)(b * SEQLEN + li)) * (2 * D_INNER) + d] * cw[d * D_CONV + j];
  }
  float s = acc / (1.f + __expf(-acc));
  xc[idx] = s;
  xcb[idx] = (__bf16)s;
}

// ---------------------------------------------------------------------------
// Extract dt_low (first DT_RANK cols of x_dbl) as bf16 for GEMM3.
// ---------------------------------------------------------------------------
__global__ __launch_bounds__(256) void extract_dtlow_kernel(
    const float* __restrict__ xdbl, __bf16* __restrict__ dtlow) {
  int idx = blockIdx.x * 256 + threadIdx.x;  // NROWS * DT_RANK
  int row = idx >> 6;
  int c = idx & (DT_RANK - 1);
  dtlow[idx] = (__bf16)xdbl[(size_t)row * XDBL_N + c];
}

// ---------------------------------------------------------------------------
// Selective scan.  16 lanes per (b,d) chain, lane = state n.
// Folds softplus(dt_raw + bias), D*x skip, SiLU(z) gate.  Emits bf16 y.
// ---------------------------------------------------------------------------
__global__ __launch_bounds__(256) void scan_kernel(
    const float* __restrict__ dtraw, const float* __restrict__ dtb,
    const float* __restrict__ Alog, const float* __restrict__ xdbl,
    const float* __restrict__ xc, const float* __restrict__ xz,
    const float* __restrict__ Dp, __bf16* __restrict__ ybf) {
  const int tid = threadIdx.x;
  const int chain = blockIdx.x * 16 + (tid >> 4);  // 0..8191
  const int n = tid & 15;
  const int b = chain >> 11;
  const int d = chain & (D_INNER - 1);

  const float An = -__expf(Alog[d * D_STATE + n]);
  const float bias = dtb[d];
  const float Dd = Dp[d];
  float h = 0.f;
  const size_t rowbase = (size_t)b * SEQLEN;

  for (int t = 0; t < SEQLEN; ++t) {
    const size_t row = rowbase + t;
    float v = dtraw[row * D_INNER + d] + bias;
    float dt = (v > 20.f) ? v : log1pf(__expf(v));
    float x = xc[row * D_INNER + d];
    float Bn = xdbl[row * XDBL_N + DT_RANK + n];
    float Cn = xdbl[row * XDBL_N + DT_RANK + D_STATE + n];
    h = __expf(dt * An) * h + (dt * x) * Bn;
    float y = h * Cn;
    y += __shfl_xor(y, 1);
    y += __shfl_xor(y, 2);
    y += __shfl_xor(y, 4);
    y += __shfl_xor(y, 8);
    if (n == 0) {
      float z = xz[row * (2 * D_INNER) + D_INNER + d];
      float g = z / (1.f + __expf(-z));
      ybf[row * D_INNER + d] = (__bf16)((y + Dd * x) * g);
    }
  }
}

// ---------------------------------------------------------------------------
// Host-side orchestration
// ---------------------------------------------------------------------------
static inline size_t align256(size_t x) { return (x + 255) & ~(size_t)255; }

extern "C" void kernel_launch(void* const* d_in, const int* in_sizes, int n_in,
                              void* d_out, int out_size, void* d_ws, size_t ws_size,
                              hipStream_t stream) {
  (void)in_sizes; (void)n_in; (void)out_size; (void)ws_size;

  const float* x         = (const float*)d_in[0];   // (4,2048,1024)
  const float* ln_gamma  = (const float*)d_in[1];
  const float* ln_beta   = (const float*)d_in[2];
  const float* in_proj_w = (const float*)d_in[3];   // (4096,1024)
  const float* conv_w    = (const float*)d_in[4];   // (2048,4)
  const float* conv_b    = (const float*)d_in[5];
  const float* x_proj_w  = (const float*)d_in[6];   // (96,2048)
  const float* dt_proj_w = (const float*)d_in[7];   // (2048,64)
  const float* dt_proj_b = (const float*)d_in[8];
  const float* A_log     = (const float*)d_in[9];   // (2048,16)
  const float* D_param   = (const float*)d_in[10];
  const float* out_proj_w= (const float*)d_in[11];  // (1024,2048)

  char* ws = (char*)d_ws;
  size_t off = 0;
  auto carve = [&](size_t bytes) { char* p = ws + off; off = align256(off + bytes); return p; };

  float*  xz       = (float*)carve((size_t)NROWS * 2 * D_INNER * 4);   // 128 MB
  __bf16* xn_bf    = (__bf16*)carve((size_t)NROWS * D_MODEL * 2);      // 16 MB
  __bf16* win_bf   = (__bf16*)carve((size_t)2 * D_INNER * D_MODEL * 2);// 8 MB
  __bf16* wxp_bf   = (__bf16*)carve((size_t)XDBL_N * D_INNER * 2);
  __bf16* wdt_bf   = (__bf16*)carve((size_t)D_INNER * DT_RANK * 2);
  __bf16* wout_bf  = (__bf16*)carve((size_t)D_MODEL * D_INNER * 2);    // 4 MB
  float*  xc_f32   = (float*)carve((size_t)NROWS * D_INNER * 4);       // 64 MB
  __bf16* xc_bf    = (__bf16*)carve((size_t)NROWS * D_INNER * 2);      // 32 MB
  float*  xdbl     = (float*)carve((size_t)NROWS * XDBL_N * 4);        // 3 MB
  __bf16* dtlow_bf = (__bf16*)carve((size_t)NROWS * DT_RANK * 2);      // 1 MB
  float*  dtraw    = (float*)carve((size_t)NROWS * D_INNER * 4);       // 64 MB
  __bf16* y_bf     = (__bf16*)carve((size_t)NROWS * D_INNER * 2);      // 32 MB

  // 1) weight conversions to bf16
  {
    int n1 = 2 * D_INNER * D_MODEL;   // 4194304
    int n2 = XDBL_N * D_INNER;        // 196608
    int n3 = D_INNER * DT_RANK;       // 131072
    int n4 = D_MODEL * D_INNER;       // 2097152
    f32_to_bf16_kernel<<<(n1 + 255) / 256, 256, 0, stream>>>(in_proj_w, win_bf, n1);
    f32_to_bf16_kernel<<<(n2 + 255) / 256, 256, 0, stream>>>(x_proj_w, wxp_bf, n2);
    f32_to_bf16_kernel<<<(n3 + 255) / 256, 256, 0, stream>>>(dt_proj_w, wdt_bf, n3);
    f32_to_bf16_kernel<<<(n4 + 255) / 256, 256, 0, stream>>>(out_proj_w, wout_bf, n4);
  }

  // 2) LayerNorm -> bf16
  layernorm_bf16_kernel<<<NROWS, 256, 0, stream>>>(x, ln_gamma, ln_beta, xn_bf);

  // 3) GEMM1: xz[8192,4096] = xn[8192,1024] @ in_proj_w^T
  {
    dim3 grid((2 * D_INNER + GBN - 1) / GBN, NROWS / GBM);
    gemm_bf16_wmma<false><<<grid, 256, 0, stream>>>(
        xn_bf, win_bf, xz, nullptr, NROWS, 2 * D_INNER, D_MODEL);
  }

  // 4) depthwise conv + SiLU -> xc (f32 + bf16)
  conv_silu_kernel<<<(NROWS * D_INNER) / 256, 256, 0, stream>>>(
      xz, conv_w, conv_b, xc_f32, xc_bf);

  // 5) GEMM2: x_dbl[8192,96] = xc[8192,2048] @ x_proj_w^T
  {
    dim3 grid((XDBL_N + GBN - 1) / GBN, NROWS / GBM);
    gemm_bf16_wmma<false><<<grid, 256, 0, stream>>>(
        xc_bf, wxp_bf, xdbl, nullptr, NROWS, XDBL_N, D_INNER);
  }

  // 6) extract dt_low -> bf16
  extract_dtlow_kernel<<<(NROWS * DT_RANK) / 256, 256, 0, stream>>>(xdbl, dtlow_bf);

  // 7) GEMM3: dt_raw[8192,2048] = dt_low[8192,64] @ dt_proj_w^T
  {
    dim3 grid((D_INNER + GBN - 1) / GBN, NROWS / GBM);
    gemm_bf16_wmma<false><<<grid, 256, 0, stream>>>(
        dtlow_bf, wdt_bf, dtraw, nullptr, NROWS, D_INNER, DT_RANK);
  }

  // 8) selective scan -> gated y (bf16)
  scan_kernel<<<(BATCH * D_INNER) / 16, 256, 0, stream>>>(
      dtraw, dt_proj_b, A_log, xdbl, xc_f32, xz, D_param, y_bf);

  // 9) GEMM4 + residual: out[8192,1024] = x + y[8192,2048] @ out_proj_w^T
  {
    dim3 grid((D_MODEL + GBN - 1) / GBN, NROWS / GBM);
    gemm_bf16_wmma<true><<<grid, 256, 0, stream>>>(
        y_bf, wout_bf, (float*)d_out, x, NROWS, D_MODEL, D_INNER);
  }
}
